// MoE_22093311771199
// MI455X (gfx1250) — matrix operand order repaired and checked by reading, not compiled, not run on hardware
//
#include <hip/hip_runtime.h>
#include <math.h>

// ---------------------------------------------------------------------------
// MoE top-1 (B=4,S=2048,H=1024,F=4096,E=8) for gfx1250 (MI455X).
//
//  1) router (wave per token) -> per-expert token bins
//  2) prep: x -> bf16, w1/w2 -> bf16 *transposed* (K-contiguous per N row)
//  3) FFN1/FFN2 grouped bf16 WMMA GEMMs whose tiles are staged with
//     GLOBAL_LOAD_ASYNC_TO_LDS_B128 (ASYNCcnt, double buffered) - no VALU
//     conversion and no LDS transpose in the inner loop.
//
// Roofline: 137 GFLOP of real routed work vs ~335MB minimum traffic -> the
// GEMMs are compute-bound at bf16 WMMA rates; prep pass is bandwidth-bound
// (~20us) and per-expert bf16 weights (8.4MB) stay resident in the 192MB L2.
// ---------------------------------------------------------------------------

typedef __attribute__((ext_vector_type(16))) __bf16 v16bf;
typedef __attribute__((ext_vector_type(8)))  __bf16 v8bf;
typedef __attribute__((ext_vector_type(4)))  __bf16 v4bf;
typedef __attribute__((ext_vector_type(8)))  float  v8f;
typedef __attribute__((ext_vector_type(4)))  float  v4f;

#define B_ 4
#define S_ 2048
#define T_ (B_ * S_)   // 8192 tokens
#define H_ 1024
#define F_ 4096
#define E_ 8

#define MT 128         // token tile
#define NT 128         // output-feature tile
#define KC 32          // K chunk (bf16 WMMA K)
#define LDA 40         // padded LDS row stride in bf16 (80B: 16B-aligned, 20
                       // words -> conflict-free across 16-lane groups)

// ---------------------------------------------------------------------------
// Router: one wave32 per token; argmax over 8 experts; atomic binning.
// ---------------------------------------------------------------------------
__global__ __launch_bounds__(256) void moe_router(
    const float* __restrict__ x, const float* __restrict__ wr,
    const float* __restrict__ br, int* __restrict__ counts,
    int* __restrict__ perm) {
  int tok  = (blockIdx.x * blockDim.x + threadIdx.x) >> 5;
  int lane = threadIdx.x & 31;
  if (tok >= T_) return;
  const float* xr = x + (size_t)tok * H_;
  float acc[E_];
#pragma unroll
  for (int e = 0; e < E_; ++e) acc[e] = 0.0f;
  for (int k = lane; k < H_; k += 32) {
    float xv = xr[k];
    const float* wrow = wr + k * E_;
#pragma unroll
    for (int e = 0; e < E_; ++e) acc[e] += xv * wrow[e];
  }
#pragma unroll
  for (int off = 16; off > 0; off >>= 1) {
#pragma unroll
    for (int e = 0; e < E_; ++e) acc[e] += __shfl_down(acc[e], off, 32);
  }
  if (lane == 0) {
    int best = 0;
    float bv = acc[0] + br[0];
#pragma unroll
    for (int e = 1; e < E_; ++e) {
      float v = acc[e] + br[e];
      if (v > bv) { bv = v; best = e; }
    }
    int pos = atomicAdd(&counts[best], 1);
    perm[best * T_ + pos] = tok;
  }
}

// ---------------------------------------------------------------------------
// Prep: fp32 -> bf16 elementwise (x).
// ---------------------------------------------------------------------------
__device__ __forceinline__ v8bf cvt8(v4f a, v4f b) {
  v8bf r;
  r[0] = (__bf16)a[0]; r[1] = (__bf16)a[1]; r[2] = (__bf16)a[2]; r[3] = (__bf16)a[3];
  r[4] = (__bf16)b[0]; r[5] = (__bf16)b[1]; r[6] = (__bf16)b[2]; r[7] = (__bf16)b[3];
  return r;
}

__global__ __launch_bounds__(256) void cvt_bf16(
    const float* __restrict__ src, __bf16* __restrict__ dst) {
  int i = blockIdx.x * 256 + threadIdx.x;   // index in units of 8 elements
  const v4f* s = (const v4f*)src + (size_t)i * 2;
  v4f a = s[0], b = s[1];
  *((v8bf*)dst + i) = cvt8(a, b);
}

// ---------------------------------------------------------------------------
// Prep: per-expert fp32 [R][C] -> bf16 transposed [C][R], 32x32 LDS tiles.
// ---------------------------------------------------------------------------
__global__ __launch_bounds__(256) void transpose_cvt(
    const float* __restrict__ src, __bf16* __restrict__ dst, int R, int C) {
  __shared__ __bf16 tile[32][36];
  int tilesR = R >> 5, tilesC = C >> 5;
  int bid = blockIdx.x;
  int e   = bid / (tilesR * tilesC);
  int rem = bid % (tilesR * tilesC);
  int br  = rem / tilesC;
  int bc  = rem % tilesC;
  int t   = threadIdx.x;
  int lr  = t >> 3, lc4 = (t & 7) * 4;

  v4f v = *(const v4f*)(src + (size_t)e * R * C + (size_t)(br * 32 + lr) * C + bc * 32 + lc4);
#pragma unroll
  for (int j = 0; j < 4; ++j) tile[lr][lc4 + j] = (__bf16)v[j];
  __syncthreads();

  int oc = t >> 3, or4 = (t & 7) * 4;   // output row = C index, cols = R index
  v4bf o;
#pragma unroll
  for (int j = 0; j < 4; ++j) o[j] = tile[or4 + j][oc];
  *(v4bf*)(dst + (size_t)e * R * C + (size_t)(bc * 32 + oc) * R + br * 32 + or4) = o;
}

// ---------------------------------------------------------------------------
// Async tile staging: 4 x global_load_async_to_lds_b128 per thread
// (A tile 128xKC + B tile 128xKC, 16B chunks, padded LDS rows).
// ---------------------------------------------------------------------------
#define ASYNC_B128(ldsaddr, voff, sbase)                                     \
  asm volatile("global_load_async_to_lds_b128 %0, %1, %2"                    \
               :: "v"(ldsaddr), "v"(voff), "s"(sbase) : "memory")

#define WAIT_ASYNC(n) asm volatile("s_wait_asynccnt " #n ::: "memory")

// ---------------------------------------------------------------------------
// Grouped GEMM mainloop shared by FFN1/FFN2 via template on K dimension.
// acc: wave grid 2(M) x 4(N), each wave 4x2 tiles of 16x16.
// ---------------------------------------------------------------------------
template <int KDIM, bool GELU>
__device__ __forceinline__ void moe_gemm(
    const __bf16* __restrict__ Abase,   // [T][KDIM] bf16, rows gathered by perm
    const __bf16* __restrict__ Bbase,   // this (e,nt)'s 128 N-rows, stride KDIM
    const float*  __restrict__ bias,    // bias for cols nt*NT..+127 (base ptr +ncol)
    const int* __restrict__ pbase, int valid, int ncol_base,
    __bf16* __restrict__ hout,          // GELU path: bf16 out [T][F]
    float*  __restrict__ fout) {        // !GELU path: fp32 out [T][H]
  __shared__ __bf16 smem[2][2][MT][LDA];   // [buf][A/B][row][k]
  const unsigned BUFB = (unsigned)(sizeof(smem) / 2);

  int tid  = threadIdx.x;
  int lane = tid & 31;
  int wv   = tid >> 5;
  int wm   = wv >> 2;     // 0..1
  int wn   = wv & 3;      // 0..3
  int kh   = lane >> 4;
  int lc   = lane & 15;

  // ---- async staging assignment: thread owns chunks (row, row+64) x 16B ----
  int q     = tid & 3;          // 16B chunk within 64B of K-row
  int rowA0 = tid >> 2;         // 0..63
  int rowA1 = rowA0 + 64;
  int ra0   = (rowA0 < valid) ? rowA0 : 0;
  int ra1   = (rowA1 < valid) ? rowA1 : 0;
  unsigned voffA0 = (unsigned)pbase[ra0] * (KDIM * 2) + q * 16;
  unsigned voffA1 = (unsigned)pbase[ra1] * (KDIM * 2) + q * 16;
  unsigned voffB0 = (unsigned)rowA0 * (KDIM * 2) + q * 16;
  unsigned voffB1 = (unsigned)rowA1 * (KDIM * 2) + q * 16;
  unsigned ldsA0 = (unsigned)(uintptr_t)&smem[0][0][rowA0][q * 8];
  unsigned ldsA1 = (unsigned)(uintptr_t)&smem[0][0][rowA1][q * 8];
  unsigned ldsB0 = (unsigned)(uintptr_t)&smem[0][1][rowA0][q * 8];
  unsigned ldsB1 = (unsigned)(uintptr_t)&smem[0][1][rowA1][q * 8];

  v8f acc[4][2];
#pragma unroll
  for (int im = 0; im < 4; ++im)
#pragma unroll
    for (int in = 0; in < 2; ++in) acc[im][in] = (v8f)0.0f;

#define ISSUE_TILES(buf, kk)                                                 \
  do {                                                                       \
    unsigned kb = (unsigned)(kk) * 2u, bo = (unsigned)(buf) * BUFB;          \
    ASYNC_B128(ldsA0 + bo, voffA0 + kb, Abase);                              \
    ASYNC_B128(ldsA1 + bo, voffA1 + kb, Abase);                              \
    ASYNC_B128(ldsB0 + bo, voffB0 + kb, Bbase);                              \
    ASYNC_B128(ldsB1 + bo, voffB1 + kb, Bbase);                              \
  } while (0)

  ISSUE_TILES(0, 0);
  for (int k0 = 0, it = 0; k0 < KDIM; k0 += KC, ++it) {
    int cb = it & 1;
    if (k0 + KC < KDIM) {
      ISSUE_TILES(cb ^ 1, k0 + KC);   // prefetch next tile into other buffer
      WAIT_ASYNC(4);                  // oldest 4 (current buffer) landed
    } else {
      WAIT_ASYNC(0);
    }
    __syncthreads();                  // all waves' chunks visible

    v16bf afr[4];
#pragma unroll
    for (int im = 0; im < 4; ++im) {
      int row = wm * 64 + im * 16 + lc;
      union { v16bf v; v8bf half[2]; } u;
      u.half[0] = *(const v8bf*)&smem[cb][0][row][kh * 8];
      u.half[1] = *(const v8bf*)&smem[cb][0][row][16 + kh * 8];
      afr[im] = u.v;
    }
    v16bf bfr[2];
#pragma unroll
    for (int in = 0; in < 2; ++in) {
      int col = wn * 32 + in * 16 + lc;
      union { v16bf v; v8bf half[2]; } u;
      u.half[0] = *(const v8bf*)&smem[cb][1][col][kh * 8];
      u.half[1] = *(const v8bf*)&smem[cb][1][col][16 + kh * 8];
      bfr[in] = u.v;
    }
#pragma unroll
    for (int im = 0; im < 4; ++im)
#pragma unroll
      for (int in = 0; in < 2; ++in)
        acc[im][in] = __builtin_amdgcn_wmma_f32_16x16x32_bf16(
            false, afr[im], false, bfr[in], (short)0, acc[im][in], false, false);

    __syncthreads();                  // reads done before buffer reuse
  }
#undef ISSUE_TILES

  // ---- epilogue: bias (+ exact gelu), scatter by token ----
#pragma unroll
  for (int im = 0; im < 4; ++im) {
#pragma unroll
    for (int in = 0; in < 2; ++in) {
      int ncol = wn * 32 + in * 16 + lc;
      float bv = bias[ncol];
#pragma unroll
      for (int r = 0; r < 8; ++r) {
        int mrow = wm * 64 + im * 16 + r + kh * 8;
        if (mrow < valid) {
          int tok = pbase[mrow];
          float v = acc[im][in][r] + bv;
          if (GELU) {
            v = 0.5f * v * (1.0f + erff(v * 0.70710678118654752440f));
            hout[(size_t)tok * F_ + ncol_base + ncol] = (__bf16)v;
          } else {
            fout[(size_t)tok * H_ + ncol_base + ncol] = v;
          }
        }
      }
    }
  }
}

// ---------------------------------------------------------------------------
__global__ __launch_bounds__(256) void moe_ffn1(
    const __bf16* __restrict__ xb, const __bf16* __restrict__ w1t,
    const float* __restrict__ b1, const int* __restrict__ counts,
    const int* __restrict__ perm, __bf16* __restrict__ h) {
  const int nt_per = F_ / NT;   // 32
  const int mt_per = T_ / MT;   // 64
  int bid = blockIdx.x;
  int e   = bid / (mt_per * nt_per);
  int rem = bid % (mt_per * nt_per);
  int mt  = rem / nt_per;
  int nt  = rem % nt_per;
  int cnt = counts[e];
  if (mt * MT >= cnt) return;
  int valid = cnt - mt * MT;
  if (valid > MT) valid = MT;
  const int* pbase = perm + e * T_ + mt * MT;

  const __bf16* Bbase = w1t + ((size_t)e * F_ + (size_t)nt * NT) * H_;
  moe_gemm<H_, true>(xb, Bbase, b1 + e * F_ + nt * NT, pbase, valid, nt * NT,
                     h, nullptr);
}

__global__ __launch_bounds__(256) void moe_ffn2(
    const __bf16* __restrict__ h, const __bf16* __restrict__ w2t,
    const float* __restrict__ b2, const int* __restrict__ counts,
    const int* __restrict__ perm, float* __restrict__ out) {
  const int nt_per = H_ / NT;   // 8
  const int mt_per = T_ / MT;   // 64
  int bid = blockIdx.x;
  int e   = bid / (mt_per * nt_per);
  int rem = bid % (mt_per * nt_per);
  int mt  = rem / nt_per;
  int nt  = rem % nt_per;
  int cnt = counts[e];
  if (mt * MT >= cnt) return;
  int valid = cnt - mt * MT;
  if (valid > MT) valid = MT;
  const int* pbase = perm + e * T_ + mt * MT;

  const __bf16* Bbase = w2t + ((size_t)e * H_ + (size_t)nt * NT) * F_;
  moe_gemm<F_, false>(h, Bbase, b2 + e * H_ + nt * NT, pbase, valid, nt * NT,
                      nullptr, out);
}

// ---------------------------------------------------------------------------
extern "C" void kernel_launch(void* const* d_in, const int* in_sizes, int n_in,
                              void* d_out, int out_size, void* d_ws, size_t ws_size,
                              hipStream_t stream) {
  (void)in_sizes; (void)n_in; (void)out_size; (void)ws_size;
  const float* x  = (const float*)d_in[0];
  const float* wr = (const float*)d_in[1];
  const float* br = (const float*)d_in[2];
  const float* w1 = (const float*)d_in[3];
  const float* b1 = (const float*)d_in[4];
  const float* w2 = (const float*)d_in[5];
  const float* b2 = (const float*)d_in[6];
  float* out = (float*)d_out;

  // workspace layout (bytes)
  char*   ws     = (char*)d_ws;
  int*    counts = (int*)ws;                              //      32 B
  int*    perm   = (int*)(ws + 256);                      //  256 KB
  size_t  off    = 262656;                                //  512-aligned
  __bf16* hbuf   = (__bf16*)(ws + off); off += (size_t)T_ * F_ * 2;       // 64 MB
  __bf16* xb     = (__bf16*)(ws + off); off += (size_t)T_ * H_ * 2;       // 16 MB
  __bf16* w1t    = (__bf16*)(ws + off); off += (size_t)E_ * H_ * F_ * 2;  // 64 MB
  __bf16* w2t    = (__bf16*)(ws + off);                                   // 64 MB

  hipMemsetAsync(counts, 0, E_ * sizeof(int), stream);

  // router: one wave per token
  moe_router<<<(T_ * 32) / 256, 256, 0, stream>>>(x, wr, br, counts, perm);

  // prep: bf16 conversion + weight transposes (K-contiguous layouts)
  cvt_bf16<<<(T_ * H_ / 8) / 256, 256, 0, stream>>>(x, xb);
  transpose_cvt<<<E_ * (H_ / 32) * (F_ / 32), 256, 0, stream>>>(w1, w1t, H_, F_);
  transpose_cvt<<<E_ * (F_ / 32) * (H_ / 32), 256, 0, stream>>>(w2, w2t, F_, H_);

  // grouped GEMMs (blocks beyond per-expert count early-exit)
  moe_ffn1<<<E_ * (T_ / MT) * (F_ / NT), 256, 0, stream>>>(xb, w1t, b1, counts, perm, hbuf);
  moe_ffn2<<<E_ * (T_ / MT) * (H_ / NT), 256, 0, stream>>>(hbuf, w2t, b2, counts, perm, out);
}
